// Block_55336358642808
// MI455X (gfx1250) — matrix operand order repaired and checked by reading, not compiled, not run on hardware
//
#include <hip/hip_runtime.h>
#include <hip/hip_bf16.h>

// ---------------------------------------------------------------------------
// Transformer block forward for gfx1250 (MI455X), f16 WMMA with f32 accum.
// B=16, S=1024, E=256, H=4, D=64.  Needs ~102MB of workspace.
// GEMMs: 128x64 block tile, LDS-staged B panel via async-to-LDS, double
// buffered; 4 WMMAs per A operand (reuse_a).  Attention: flash-style causal,
// V panel async-staged to LDS per wave.
// ---------------------------------------------------------------------------

#define BATCH 16
#define SEQ   1024
#define EMBD  256
#define NHEAD 4
#define HD    64
#define NTOK  (BATCH * SEQ)   // 16384
#define EPS   1e-5f

typedef _Float16 v16h __attribute__((ext_vector_type(16)));
typedef _Float16 v8h  __attribute__((ext_vector_type(8)));
typedef float    v8f  __attribute__((ext_vector_type(8)));

union H16 { v16h v; v8h h[2]; _Float16 e[16]; };

template <bool RA>
static __device__ __forceinline__ v8f wmma_f16(v16h a, v16h b, v8f c) {
  // D = A(16x32 f16) * B(32x16 f16) + C(16x16 f32)
  return __builtin_amdgcn_wmma_f32_16x16x32_f16(
      /*neg_a=*/false, a, /*neg_b=*/false, b,
      /*c_mod=*/(short)0, c, /*reuse_a=*/RA, /*reuse_b=*/false);
}

// Async global -> LDS copy of 16 bytes (tracked by ASYNCcnt).
static __device__ __forceinline__ void async_b128(unsigned lds_off,
                                                  const void* gaddr) {
  asm volatile("global_load_async_to_lds_b128 %0, %1, off"
               :: "v"(lds_off), "v"((unsigned long long)(uintptr_t)gaddr)
               : "memory");
}
static __device__ __forceinline__ void wait_async0() {
  asm volatile("s_wait_asynccnt 0x0" ::: "memory");
}
static __device__ __forceinline__ void wait_async1() {
  asm volatile("s_wait_asynccnt 0x1" ::: "memory");
}

// ---------------------------------------------------------------------------
// Weight convert + transpose: w[K,N] f32 -> wt[N,K] f16
// ---------------------------------------------------------------------------
__global__ __launch_bounds__(256)
void cvt_transpose(const float* __restrict__ w, _Float16* __restrict__ wt,
                   int K, int N) {
  int idx = blockIdx.x * 256 + threadIdx.x;
  if (idx >= K * N) return;
  int k = idx / N;
  int n = idx % N;
  wt[(size_t)n * K + k] = (_Float16)w[idx];
}

// ---------------------------------------------------------------------------
// LayerNorm over last dim (256). One wave per row, 8 rows per block.
// ---------------------------------------------------------------------------
__global__ __launch_bounds__(256)
void ln_kernel(const float* __restrict__ x, const float* __restrict__ g,
               const float* __restrict__ bta, _Float16* __restrict__ out,
               int rows) {
  int wid  = threadIdx.x >> 5;
  int lane = threadIdx.x & 31;
  int row  = blockIdx.x * 8 + wid;
  if (row >= rows) return;
  const float* xr = x + (size_t)row * EMBD + lane * 8;
  float v[8];
  #pragma unroll
  for (int j = 0; j < 8; ++j) v[j] = xr[j];
  float s = 0.f, s2 = 0.f;
  #pragma unroll
  for (int j = 0; j < 8; ++j) { s += v[j]; s2 += v[j] * v[j]; }
  #pragma unroll
  for (int m = 1; m < 32; m <<= 1) {
    s  += __shfl_xor(s,  m);
    s2 += __shfl_xor(s2, m);
  }
  float mu  = s * (1.0f / EMBD);
  float var = s2 * (1.0f / EMBD) - mu * mu;
  float rs  = rsqrtf(var + EPS);
  _Float16* orow = out + (size_t)row * EMBD + lane * 8;
  #pragma unroll
  for (int j = 0; j < 8; ++j) {
    int col = lane * 8 + j;
    orow[j] = (_Float16)((v[j] - mu) * rs * g[col] + bta[col]);
  }
}

// ---------------------------------------------------------------------------
// WMMA GEMM:  C[M,N] = A[M,K](f16) * Bt[N,K](f16)^T + bias
//             (+ residual f32)  (+ ReLU)  -> f16 or f32 output.
// Block tile 128x64: 8 waves, each 16 rows x 64 cols (4 accumulators).
// B panel (64x32 per K-chunk) staged in LDS by async-to-LDS, double buffered.
// Grid must exactly cover M/128 x N/64 (true for all uses here).
// ---------------------------------------------------------------------------
template <bool OUTF16, bool RELU, bool HASRES>
__global__ __launch_bounds__(256)
void gemm_wmma(const _Float16* __restrict__ A, const _Float16* __restrict__ Bt,
               const float* __restrict__ bias, const float* __restrict__ res,
               void* __restrict__ C, int M, int N, int K) {
  __shared__ _Float16 sB[2][64][32];   // [buffer][n-local][k-chunk]

  int wid  = threadIdx.x >> 5;
  int lane = threadIdx.x & 31;
  int lcol = lane & 15;
  int bn_cnt = N >> 6;
  int bm = blockIdx.x / bn_cnt;
  int bn = blockIdx.x % bn_cnt;
  int tm = (bm << 7) + (wid << 4);     // this wave's 16 rows

  int aoff = (lane < 16) ? 0 : 16;     // A operand: sequential K per half-wave
  int boff = (lane < 16) ? 0 : 16;     // B operand: sequential K per half-wave

  const _Float16* arow = A + (size_t)(tm + lcol) * K;

  // staging assignment: thread -> (n-local, k-part) 16B chunk
  int nl = threadIdx.x >> 2;           // 0..63
  int kp = (threadIdx.x & 3) << 3;     // 0,8,16,24
  const _Float16* gB = Bt + (size_t)((bn << 6) + nl) * K + kp;
  unsigned ldsB[2];
  ldsB[0] = (unsigned)(uintptr_t)&sB[0][nl][kp];
  ldsB[1] = (unsigned)(uintptr_t)&sB[1][nl][kp];

  v8f acc0 = {}, acc1 = {}, acc2 = {}, acc3 = {};
  int nchunks = K >> 5;

  async_b128(ldsB[0], gB);             // prologue: chunk 0 -> buffer 0
  for (int i = 0; i < nchunks; ++i) {
    int cur = i & 1;
    if (i + 1 < nchunks) {
      async_b128(ldsB[cur ^ 1], gB + ((i + 1) << 5));
      wait_async1();                   // chunk i complete (in-order)
    } else {
      wait_async0();
    }
    __syncthreads();                   // all threads' panel pieces visible

    H16 a;
    const _Float16* ap = arow + (i << 5) + aoff;
    a.h[0] = *(const v8h*)ap;
    a.h[1] = *(const v8h*)(ap + 8);
    v16h b0 = *(const v16h*)&sB[cur][lcol][boff];
    v16h b1 = *(const v16h*)&sB[cur][16 + lcol][boff];
    v16h b2 = *(const v16h*)&sB[cur][32 + lcol][boff];
    v16h b3 = *(const v16h*)&sB[cur][48 + lcol][boff];
    acc0 = wmma_f16<true >(a.v, b0, acc0);
    acc1 = wmma_f16<true >(a.v, b1, acc1);
    acc2 = wmma_f16<true >(a.v, b2, acc2);
    acc3 = wmma_f16<false>(a.v, b3, acc3);

    __syncthreads();                   // reads done before buffer reuse
  }

  int rhalf = (lane >= 16) ? 8 : 0;
  v8f accs[4] = {acc0, acc1, acc2, acc3};
  #pragma unroll
  for (int j = 0; j < 4; ++j) {
    int col = (bn << 6) + (j << 4) + lcol;
    float bcol = bias[col];
    #pragma unroll
    for (int r = 0; r < 8; ++r) {
      int row = tm + r + rhalf;
      float v = accs[j][r] + bcol;
      if (HASRES) v += res[(size_t)row * N + col];
      if (RELU)   v = fmaxf(v, 0.0f);
      if (OUTF16) ((_Float16*)C)[(size_t)row * N + col] = (_Float16)v;
      else        ((float*)C)[(size_t)row * N + col] = v;
    }
  }
}

// ---------------------------------------------------------------------------
// Causal flash attention. qkv: [NTOK, 768] f16 (Q|K|V, head-major inside).
// One wave per (batch, head, 16-query block); Bc = 32 keys.
// V panel (32x64) async-staged to this wave's LDS region each key block,
// overlapped with the QK^T + softmax work.
// ---------------------------------------------------------------------------
__global__ __launch_bounds__(256)
void attn_kernel(const _Float16* __restrict__ qkv, _Float16* __restrict__ o) {
  __shared__ _Float16 pl[8][16][32];   // P staging (C-layout -> A-layout)
  __shared__ _Float16 sV[8][32][64];   // V panel per wave

  int wid  = threadIdx.x >> 5;
  int lane = threadIdx.x & 31;
  int w    = blockIdx.x * 8 + wid;     // 0 .. 4095
  int qb = w & 63;                     // query block (S/16 = 64)
  int h  = (w >> 6) & 3;
  int b  = w >> 8;
  int q0 = qb << 4;

  const _Float16* base = qkv + (size_t)b * SEQ * 768;
  int aoff  = (lane < 16) ? 0 : 16;
  int boff  = (lane < 16) ? 0 : 16;
  int rhalf = (lane >= 16) ? 8 : 0;
  int lcol  = lane & 15;

  // Q operand (two K-chunks of 32 over d=0..63)
  const _Float16* qrow = base + (size_t)(q0 + lcol) * 768 + h * HD;
  H16 aq0, aq1;
  aq0.h[0] = *(const v8h*)(qrow + aoff);
  aq0.h[1] = *(const v8h*)(qrow + aoff + 8);
  aq1.h[0] = *(const v8h*)(qrow + 32 + aoff);
  aq1.h[1] = *(const v8h*)(qrow + 32 + aoff + 8);

  float mrow[8], lrow[8];
  v8f acc[4];
  #pragma unroll
  for (int r = 0; r < 8; ++r) { mrow[r] = -1e30f; lrow[r] = 0.0f; }
  #pragma unroll
  for (int t = 0; t < 4; ++t) acc[t] = (v8f){};

  unsigned ldsV = (unsigned)(uintptr_t)&sV[wid][lane][0];

  for (int k0 = 0; k0 < q0 + 16; k0 += 32) {
    // ---- kick off async V panel copy: lane 'lane' fetches V row (k0+lane)
    const _Float16* vrow = base + (size_t)(k0 + lane) * 768 + 2 * EMBD + h * HD;
    #pragma unroll
    for (int i = 0; i < 8; ++i)
      async_b128(ldsV + i * 16, vrow + i * 8);

    // ---- S = Q K^T * scale : two 16x16 score tiles
    const _Float16* kp0 = base + (size_t)(k0 + lcol) * 768 + EMBD + h * HD;
    const _Float16* kp1 = kp0 + 16 * 768;
    v8f s0 = {}, s1 = {};
    s0 = wmma_f16<false>(aq0.v, *(const v16h*)(kp0 + boff),      s0);
    s0 = wmma_f16<false>(aq1.v, *(const v16h*)(kp0 + 32 + boff), s0);
    s1 = wmma_f16<false>(aq0.v, *(const v16h*)(kp1 + boff),      s1);
    s1 = wmma_f16<false>(aq1.v, *(const v16h*)(kp1 + 32 + boff), s1);

    int key0 = k0 + lcol;
    int key1 = key0 + 16;
    float p0[8], p1[8], t8[8];
    #pragma unroll
    for (int r = 0; r < 8; ++r) {
      int qg = q0 + r + rhalf;
      float a  = s0[r] * 0.125f;        // 1/sqrt(64)
      float bb = s1[r] * 0.125f;
      if (key0 > qg) a  = -1e30f;
      if (key1 > qg) bb = -1e30f;
      p0[r] = a; p1[r] = bb;
      t8[r] = fmaxf(a, bb);
    }
    #pragma unroll
    for (int m = 1; m < 16; m <<= 1)
      #pragma unroll
      for (int r = 0; r < 8; ++r) t8[r] = fmaxf(t8[r], __shfl_xor(t8[r], m));

    float alpha[8];
    #pragma unroll
    for (int r = 0; r < 8; ++r) {
      float mn = fmaxf(mrow[r], t8[r]);
      alpha[r] = __expf(mrow[r] - mn);
      mrow[r] = mn;
      p0[r] = __expf(p0[r] - mn);
      p1[r] = __expf(p1[r] - mn);
      t8[r] = p0[r] + p1[r];
    }
    #pragma unroll
    for (int m = 1; m < 16; m <<= 1)
      #pragma unroll
      for (int r = 0; r < 8; ++r) t8[r] += __shfl_xor(t8[r], m);
    #pragma unroll
    for (int r = 0; r < 8; ++r) {
      lrow[r] = lrow[r] * alpha[r] + t8[r];
      #pragma unroll
      for (int t = 0; t < 4; ++t) acc[t][r] *= alpha[r];
    }

    // ---- stage P (C-layout) into LDS, reload in A-operand layout
    #pragma unroll
    for (int r = 0; r < 8; ++r) {
      pl[wid][r + rhalf][lcol]      = (_Float16)p0[r];
      pl[wid][r + rhalf][lcol + 16] = (_Float16)p1[r];
    }
    H16 pa;
    pa.h[0] = *(const v8h*)(&pl[wid][lcol][aoff]);
    pa.h[1] = *(const v8h*)(&pl[wid][lcol][aoff + 8]);

    // ---- V panel ready; gather transposed B operands from LDS
    wait_async0();
    #pragma unroll
    for (int t = 0; t < 4; ++t) {
      int d = (t << 4) + lcol;
      H16 bv;
      #pragma unroll
      for (int i = 0; i < 16; ++i) bv.e[i] = sV[wid][boff + i][d];
      acc[t] = wmma_f16<false>(pa.v, bv.v, acc[t]);
    }
  }

  // ---- normalize + store O [token, h*64+d] f16
  _Float16* ob = o + (size_t)(b * SEQ + q0) * EMBD + h * HD;
  #pragma unroll
  for (int r = 0; r < 8; ++r) {
    float inv = 1.0f / lrow[r];
    #pragma unroll
    for (int t = 0; t < 4; ++t) {
      ob[(size_t)(r + rhalf) * EMBD + (t << 4) + lcol] =
          (_Float16)(acc[t][r] * inv);
    }
  }
}

// ---------------------------------------------------------------------------
// Host-side launch (graph-capture safe: kernels only, all on `stream`).
// ---------------------------------------------------------------------------
extern "C" void kernel_launch(void* const* d_in, const int* in_sizes, int n_in,
                              void* d_out, int out_size, void* d_ws, size_t ws_size,
                              hipStream_t stream) {
  const float* x      = (const float*)d_in[0];
  const float* qkv_w  = (const float*)d_in[1];
  const float* qkv_b  = (const float*)d_in[2];
  const float* out_w  = (const float*)d_in[3];
  const float* out_b  = (const float*)d_in[4];
  const float* fc_w   = (const float*)d_in[5];
  const float* fc_b   = (const float*)d_in[6];
  const float* proj_w = (const float*)d_in[7];
  const float* proj_b = (const float*)d_in[8];
  const float* ln1_g  = (const float*)d_in[9];
  const float* ln1_b  = (const float*)d_in[10];
  const float* ln2_g  = (const float*)d_in[11];
  const float* ln2_b  = (const float*)d_in[12];

  char* ws = (char*)d_ws;
  size_t off = 0;
  auto salloc = [&](size_t bytes) -> void* {
    void* p = ws + off;
    off += (bytes + 255) & ~(size_t)255;
    return p;
  };
  _Float16* h1    = (_Float16*)salloc((size_t)NTOK * EMBD * 2);       // 8 MB
  _Float16* qkvf  = (_Float16*)salloc((size_t)NTOK * 3 * EMBD * 2);   // 24 MB
  _Float16* of16  = (_Float16*)salloc((size_t)NTOK * EMBD * 2);       // 8 MB
  float*    x1    = (float*)   salloc((size_t)NTOK * EMBD * 4);       // 16 MB
  _Float16* h2    = (_Float16*)salloc((size_t)NTOK * EMBD * 2);       // 8 MB
  _Float16* fcf   = (_Float16*)salloc((size_t)NTOK * 4 * EMBD * 2);   // 32 MB
  _Float16* wq_t  = (_Float16*)salloc((size_t)EMBD * 3 * EMBD * 2);
  _Float16* wo_t  = (_Float16*)salloc((size_t)EMBD * EMBD * 2);
  _Float16* wfc_t = (_Float16*)salloc((size_t)EMBD * 4 * EMBD * 2);
  _Float16* wpj_t = (_Float16*)salloc((size_t)4 * EMBD * EMBD * 2);
  (void)ws_size; (void)in_sizes; (void)n_in; (void)out_size;

  // 0) weight convert + transpose
  cvt_transpose<<<(EMBD * 3 * EMBD + 255) / 256, 256, 0, stream>>>(qkv_w, wq_t, EMBD, 3 * EMBD);
  cvt_transpose<<<(EMBD * EMBD + 255) / 256, 256, 0, stream>>>(out_w, wo_t, EMBD, EMBD);
  cvt_transpose<<<(EMBD * 4 * EMBD + 255) / 256, 256, 0, stream>>>(fc_w, wfc_t, EMBD, 4 * EMBD);
  cvt_transpose<<<(4 * EMBD * EMBD + 255) / 256, 256, 0, stream>>>(proj_w, wpj_t, 4 * EMBD, EMBD);

  // 1) h1 = LN1(x)
  ln_kernel<<<NTOK / 8, 256, 0, stream>>>(x, ln1_g, ln1_b, h1, NTOK);

  // 2) qkv = h1 @ qkv_w + qkv_b   (f16 out)   grid: (M/128)*(N/64)
  gemm_wmma<true, false, false><<<(NTOK / 128) * (3 * EMBD / 64), 256, 0, stream>>>(
      h1, wq_t, qkv_b, nullptr, qkvf, NTOK, 3 * EMBD, EMBD);

  // 3) causal attention -> of16
  attn_kernel<<<(BATCH * NHEAD * (SEQ / 16)) / 8, 256, 0, stream>>>(qkvf, of16);

  // 4) x1 = x + O @ out_w + out_b   (f32 out)
  gemm_wmma<false, false, true><<<(NTOK / 128) * (EMBD / 64), 256, 0, stream>>>(
      of16, wo_t, out_b, x, x1, NTOK, EMBD, EMBD);

  // 5) h2 = LN2(x1)
  ln_kernel<<<NTOK / 8, 256, 0, stream>>>(x1, ln2_g, ln2_b, h2, NTOK);

  // 6) fcf = relu(h2 @ fc_w + fc_b)   (f16 out)
  gemm_wmma<true, true, false><<<(NTOK / 128) * (4 * EMBD / 64), 256, 0, stream>>>(
      h2, wfc_t, fc_b, nullptr, fcf, NTOK, 4 * EMBD, EMBD);

  // 7) out = x1 + fcf @ proj_w + proj_b   (f32 out -> d_out)
  gemm_wmma<false, false, true><<<(NTOK / 128) * (EMBD / 64), 256, 0, stream>>>(
      fcf, wpj_t, proj_b, x1, (float*)d_out, NTOK, EMBD, 4 * EMBD);
}
